// TestImageModel_71004399338074
// MI455X (gfx1250) — matrix operand order repaired and checked by reading, not compiled, not run on hardware
//
#include <hip/hip_runtime.h>
#include <math.h>

// ---------------- model constants ----------------
#define BSZ   64
#define NPER0 2048
#define ETOT  1048576
#define INDIM 128
#define DM    256
#define HEADS 4
#define HID   64
#define K1C   1024
#define K2C   512
#define MFEAT 128
#define DH    64
#define N0    (BSZ*NPER0)   // 131072
#define N1    (BSZ*K1C)     // 65536
#define N2    (BSZ*K2C)     // 32768

typedef __attribute__((ext_vector_type(16))) _Float16 v16h;
typedef __attribute__((ext_vector_type(8)))  float    v8f;
typedef __attribute__((ext_vector_type(4)))  unsigned u32x4;
typedef __attribute__((ext_vector_type(8)))  int      i32x8;
typedef __attribute__((ext_vector_type(4)))  int      i32x4;

// order-preserving float<->uint key (for atomic max on floats incl. negatives)
__host__ __device__ inline unsigned f2o(float f) {
  unsigned u;
#ifdef __HIP_DEVICE_COMPILE__
  u = __float_as_uint(f);
#else
  union { float f; unsigned u; } c; c.f = f; u = c.u;
#endif
  return (u & 0x80000000u) ? ~u : (u | 0x80000000u);
}
__device__ inline float o2f(unsigned k) {
  unsigned u = (k & 0x80000000u) ? (k ^ 0x80000000u) : ~k;
  return __uint_as_float(u);
}

// ================= generic batched WMMA GEMM =================
// C[bo,bi] = alpha*A[bo,bi] x B[bo,bi] (+bias[col]) (+resid) (relu?)
// A is MxK (or KxM if transA), B is KxN (or NxK if transB), C is MxN row-major.
// B tile is DMA'd into LDS by the Tensor Data Mover (non-transposed path);
// A tile is staged+converted to f16 by the VALU concurrently.
#define BM 64
#define BN 128
#define BK 64
#define BS_STRIDE 129   // 128 cols + 1 pad dword (TDM pad_interval=128dw, pad_amount=1dw)

struct GemmP {
  const float* A; const float* B; float* C;
  const float* bias; const float* resid;
  int M, N, K, lda, ldb, ldc;
  long sAo, sAi, sBo, sBi, sCo, sCi;
  int innerB, transA, transB, relu;
  float alpha;
};

__global__ __launch_bounds__(256) void gemm_wmma(GemmP p) {
  const int bz = blockIdx.z;
  const int bo = bz / p.innerB, bi = bz % p.innerB;
  const float* A = p.A + (long)bo * p.sAo + (long)bi * p.sAi;
  const float* B = p.B + (long)bo * p.sBo + (long)bi * p.sBi;
  float*       C = p.C + (long)bo * p.sCo + (long)bi * p.sCi;

  __shared__ _Float16 As[BM][BK + 4];
  __shared__ float    Bs[BK * BS_STRIDE];

  const int tid  = threadIdx.x;
  const int wave = tid >> 5, lane = tid & 31;
  const int wm = wave >> 2, wn = wave & 3;       // 2 x 4 wave grid
  const int l15 = lane & 15, hi = lane >> 4;
  const int m0 = blockIdx.y * BM + wm * 32;
  const int n0 = blockIdx.x * BN + wn * 32;

  v8f acc[2][2] = {};

  for (int kb = 0; kb < p.K; kb += BK) {
    // ---- B tile -> LDS ----
    if (!p.transB) {
      // Tensor Data Mover: 2D tile (BN elems x BK rows), f32, row stride ldb,
      // +1 dword LDS pad per 128-dword row. Issued once per block (wave 0).
      if (wave == 0) {
        unsigned ldsa = (unsigned)(size_t)(&Bs[0]);
        const float* gB = B + (long)kb * p.ldb + (long)blockIdx.x * BN;
        unsigned long long ga = (unsigned long long)(size_t)gB;
        int remN = p.N - blockIdx.x * BN;
        if (remN > BN) remN = BN;
        if (remN < 0)  remN = 0;
        int remK = p.K - kb;
        if (remK > BK) remK = BK;
        u32x4 g0;
        g0[0] = 1u;                                        // count=1 (valid user D#)
        g0[1] = ldsa;                                      // lds_addr
        g0[2] = (unsigned)ga;                              // global_addr[31:0]
        g0[3] = (unsigned)((ga >> 32) & 0x01FFFFFFu)       // global_addr[56:32]
              | (2u << 30);                                // type=2 ("image")
        i32x8 g1;
        unsigned td0 = (unsigned)remN;                     // tensor_dim0 (elems)
        unsigned td1 = (unsigned)remK;                     // tensor_dim1 (rows)
        g1[0] = (int)((2u << 16)                           // data_size = 4B
                    | (1u << 20)                           // pad_enable
                    | (6u << 22)                           // pad_interval = 128 dw
                    | (0u << 25));                         // pad_amount = 1 dw
        g1[1] = (int)((td0 & 0xFFFFu) << 16);              // tensor_dim0[15:0]
        g1[2] = (int)(((td0 >> 16) & 0xFFFFu)              // tensor_dim0[31:16]
                    | ((td1 & 0xFFFFu) << 16));            // tensor_dim1[15:0]
        g1[3] = (int)(((td1 >> 16) & 0xFFFFu)              // tensor_dim1[31:16]
                    | ((unsigned)BN << 16));               // tile_dim0 = 128
        g1[4] = (int)(unsigned)BK;                         // tile_dim1 = 64 (tile_dim2=0)
        g1[5] = (int)(unsigned)p.ldb;                      // tensor_dim0_stride[31:0]
        g1[6] = 0;                                         // stride[47:32] | dim1_stride lo
        g1[7] = 0;
        i32x4 gz4 = {0, 0, 0, 0};
        i32x8 gz8 = {0, 0, 0, 0, 0, 0, 0, 0};
        __builtin_amdgcn_tensor_load_to_lds(g0, g1, gz4, gz4, gz8, 0);
      }
    } else {
      // transposed B (e.g. omega^T): manual staging
      for (int i = tid; i < BK * BN; i += 256) {
        int k = i >> 7, n = i & 127;
        int gk = kb + k, gn = blockIdx.x * BN + n;
        float val = 0.f;
        if (gn < p.N && gk < p.K) val = B[(long)gn * p.ldb + gk];
        Bs[k * BS_STRIDE + n] = val;
      }
    }
    // ---- A tile -> LDS (f32 -> f16, alpha folded) ----
    for (int i = tid; i < BM * BK; i += 256) {
      int m = i >> 6, k = i & 63;
      int gm = blockIdx.y * BM + m, gk = kb + k;
      float val = 0.f;
      if (gm < p.M && gk < p.K)
        val = p.transA ? A[(long)gk * p.lda + gm] : A[(long)gm * p.lda + gk];
      As[m][k] = (_Float16)(val * p.alpha);
    }
    // prefetch next A rows (speculative; global_prefetch_b8)
    if (kb + BK < p.K && tid < 64) {
      int gm = blockIdx.y * BM + tid;
      if (gm < p.M) {
        long aoff = p.transA ? ((long)(kb + BK) * p.lda + gm)
                             : ((long)gm * p.lda + (kb + BK));
        __builtin_prefetch(A + aoff, 0, 1);
      }
    }
    if (!p.transB && wave == 0)
      __builtin_amdgcn_s_wait_tensorcnt(0);
    __syncthreads();

    #pragma unroll
    for (int kk = 0; kk < 2; ++kk) {
      v16h af[2], bf[2];
      #pragma unroll
      for (int t = 0; t < 2; ++t) {
        const int mA = wm * 32 + t * 16 + l15;
        #pragma unroll
        for (int vv = 0; vv < 8; ++vv) {
          // 16-bit A 16x32 layout: lanes0-15 K=0..7,16..23 ; lanes16-31 K=8..15,24..31
          int k0 = kk * 32 + ((vv < 4) ? 2 * vv : 16 + 2 * (vv - 4)) + 8 * hi;
          af[t][2 * vv]     = As[mA][k0];
          af[t][2 * vv + 1] = As[mA][k0 + 1];
        }
        const int nB = wn * 32 + t * 16 + l15;
        #pragma unroll
        for (int vv = 0; vv < 8; ++vv) {
          // 16-bit B 32x16 layout: lanes0-15 K=0..7,16..23 ; lanes16-31 K=8..15,24..31
          int k0 = kk * 32 + ((vv < 4) ? 2 * vv : 16 + 2 * (vv - 4)) + 8 * hi;
          bf[t][2 * vv]     = (_Float16)Bs[k0 * BS_STRIDE + nB];
          bf[t][2 * vv + 1] = (_Float16)Bs[(k0 + 1) * BS_STRIDE + nB];
        }
      }
      #pragma unroll
      for (int i = 0; i < 2; ++i)
        #pragma unroll
        for (int j = 0; j < 2; ++j)
          acc[i][j] = __builtin_amdgcn_wmma_f32_16x16x32_f16(
              false, af[i], false, bf[j], (short)0, acc[i][j], false, false);
    }
    __syncthreads();
  }

  // epilogue
  #pragma unroll
  for (int i = 0; i < 2; ++i) {
    #pragma unroll
    for (int j = 0; j < 2; ++j) {
      const int col = n0 + j * 16 + l15;
      if (col >= p.N) continue;
      const float bv = p.bias ? p.bias[col] : 0.f;
      #pragma unroll
      for (int r = 0; r < 8; ++r) {
        int row = m0 + i * 16 + r + 8 * hi;
        if (row < p.M) {
          float val = acc[i][j][r] + bv;
          if (p.resid) val += p.resid[(long)row * p.ldc + col];
          if (p.relu) val = val > 0.f ? val : 0.f;
          C[(long)row * p.ldc + col] = val;
        }
      }
    }
  }
}

// ================= elementwise / graph kernels =================

__global__ void fill_f32(float* p, float v, long n) {
  long i = (long)blockIdx.x * blockDim.x + threadIdx.x;
  if (i < n) p[i] = v;
}
__global__ void fill_u32(unsigned* p, unsigned v, long n) {
  long i = (long)blockIdx.x * blockDim.x + threadIdx.x;
  if (i < n) p[i] = v;
}
__global__ void bias_init(float* out, const float* bias, long rows) {
  long i = (long)blockIdx.x * blockDim.x + threadIdx.x;
  if (i < rows * DM) out[i] = bias[i & (DM - 1)];
}
__global__ void add2(float* out, const float* a, const float* b, long n) {
  long i = (long)blockIdx.x * blockDim.x + threadIdx.x;
  if (i < n) out[i] = a[i] + b[i];
}

// per-node per-head attention coefficients
__global__ void attn_coef(const float* xw, const float* as, const float* ad,
                          float* es, float* ed, int n) {
  int i = blockIdx.x * blockDim.x + threadIdx.x;
  if (i >= n * HEADS) return;
  int node = i >> 2, h = i & 3;
  const float* row = xw + (long)node * DM + h * HID;
  float s = 0.f, d = 0.f;
  for (int j = 0; j < HID; ++j) { s += row[j] * as[h * HID + j]; d += row[j] * ad[h * HID + j]; }
  es[i] = s; ed[i] = d;
}

__device__ inline void edge_sd(int idx, const int* src, const int* dst,
                               const unsigned char* mask, int E_, int& s, int& d, bool& mk) {
  if (idx < E_) { s = src[idx]; d = dst[idx]; mk = mask ? (mask[idx] != 0) : true; }
  else          { s = idx - E_; d = s; mk = true; }   // self loops
}

__global__ void edge_amax(const int* src, const int* dst, const unsigned char* mask,
                          const float* es, const float* ed, unsigned* amax, int E_, int n) {
  int idx = blockIdx.x * blockDim.x + threadIdx.x;
  if (idx >= E_ + n) return;
  int s, d; bool mk; edge_sd(idx, src, dst, mask, E_, s, d, mk);
  if (!mk) return;
  #pragma unroll
  for (int h = 0; h < HEADS; ++h) {
    float a = es[s * HEADS + h] + ed[d * HEADS + h];
    a = a > 0.f ? a : 0.2f * a;
    atomicMax(&amax[d * HEADS + h], f2o(a));
  }
}

__global__ void edge_expsum(const int* src, const int* dst, const unsigned char* mask,
                            const float* es, const float* ed, const unsigned* amax,
                            float* den, int E_, int n) {
  int idx = blockIdx.x * blockDim.x + threadIdx.x;
  if (idx >= E_ + n) return;
  int s, d; bool mk; edge_sd(idx, src, dst, mask, E_, s, d, mk);
  if (!mk) return;
  #pragma unroll
  for (int h = 0; h < HEADS; ++h) {
    float a = es[s * HEADS + h] + ed[d * HEADS + h];
    a = a > 0.f ? a : 0.2f * a;
    atomicAdd(&den[d * HEADS + h], __expf(a - o2f(amax[d * HEADS + h])));
  }
}

// wave per edge; lane strides over 256 columns
__global__ void edge_agg(const int* src, const int* dst, const unsigned char* mask,
                         const float* es, const float* ed, const unsigned* amax,
                         const float* den, const float* xw, float* out, int E_, int n) {
  int gw   = (blockIdx.x * blockDim.x + threadIdx.x) >> 5;
  int lane = threadIdx.x & 31;
  if (gw >= E_ + n) return;
  int s, d; bool mk; edge_sd(gw, src, dst, mask, E_, s, d, mk);
  if (!mk) return;
  float alpha[HEADS];
  #pragma unroll
  for (int h = 0; h < HEADS; ++h) {
    float a = es[s * HEADS + h] + ed[d * HEADS + h];
    a = a > 0.f ? a : 0.2f * a;
    float ae = __expf(a - o2f(amax[d * HEADS + h]));
    float dn = den[d * HEADS + h]; dn = dn > 1e-16f ? dn : 1e-16f;
    alpha[h] = ae / dn;
  }
  for (int c = lane; c < DM; c += 32)
    atomicAdd(&out[(long)d * DM + c], xw[(long)s * DM + c] * alpha[c >> 6]);
}

// ---------------- TopK pooling ----------------
__global__ void vec_norm(const float* p, float* out, int n) {
  __shared__ float red[256];
  float s = 0.f;
  for (int i = threadIdx.x; i < n; i += 256) s += p[i] * p[i];
  red[threadIdx.x] = s; __syncthreads();
  for (int st = 128; st > 0; st >>= 1) {
    if (threadIdx.x < st) red[threadIdx.x] += red[threadIdx.x + st];
    __syncthreads();
  }
  if (threadIdx.x == 0) out[0] = sqrtf(red[0]) + 1e-16f;
}

__global__ void score_k(const float* h, const float* p, const float* nrm, float* sc, int n) {
  int i = blockIdx.x * blockDim.x + threadIdx.x;
  if (i >= n) return;
  const float* r = h + (long)i * DM;
  float s = 0.f;
  for (int c = 0; c < DM; ++c) s += r[c] * p[c];
  sc[i] = s / nrm[0];
}

// per-graph bitonic sort (descending, index-ascending tiebreak = lax.top_k)
__global__ void topk_sort(const float* score, float* vals, int* perm, int* newid,
                          int NP, int k) {
  __shared__ float key[2048];
  __shared__ int   id[2048];
  int b = blockIdx.x;
  const float* sc = score + (long)b * NP;
  for (int i = threadIdx.x; i < NP; i += blockDim.x) { key[i] = sc[i]; id[i] = i; }
  for (int size = 2; size <= NP; size <<= 1) {
    for (int stride = size >> 1; stride > 0; stride >>= 1) {
      __syncthreads();
      for (int i = threadIdx.x; i < NP; i += blockDim.x) {
        int j = i ^ stride;
        if (j > i && j < NP) {
          float ka = key[i], kb = key[j];
          int   ia = id[i],  ib = id[j];
          bool aGreater = (ka > kb) || (ka == kb && ia < ib);
          bool up = ((i & size) == 0);
          if (up ? !aGreater : aGreater) {
            key[i] = kb; key[j] = ka; id[i] = ib; id[j] = ia;
          }
        }
      }
    }
  }
  __syncthreads();
  for (int i = threadIdx.x; i < NP; i += blockDim.x) newid[(long)b * NP + i] = -1;
  __syncthreads();
  for (int i = threadIdx.x; i < k; i += blockDim.x) {
    vals[(long)b * k + i] = key[i];
    perm[(long)b * k + i] = id[i];
    newid[(long)b * NP + id[i]] = i;
  }
}

__global__ void topk_gather(const float* h, const float* vals, const int* perm,
                            float* out, int NP, int k, int relu) {
  long i = (long)blockIdx.x * blockDim.x + threadIdx.x;
  if (i >= (long)BSZ * k * DM) return;
  int  c   = (int)(i & (DM - 1));
  long row = i >> 8;
  int  b   = (int)(row / k), r = (int)(row % k);
  float t = tanhf(vals[row]);
  float v = h[((long)b * NP + perm[row]) * DM + c] * t;
  if (relu) v = v > 0.f ? v : 0.f;
  out[i] = v;
}

__global__ void edge_remap(const int* src, const int* dst, const unsigned char* mask,
                           const int* newid, int NP, int k,
                           int* ns, int* nd, unsigned char* nm, int E_) {
  int e = blockIdx.x * blockDim.x + threadIdx.x;
  if (e >= E_) return;
  int s = src[e], d = dst[e];
  bool mk = mask ? (mask[e] != 0) : true;
  int bs = s / NP, ls = s % NP, bd = d / NP, ld = d % NP;
  int a = newid[(long)bs * NP + ls], bb = newid[(long)bd * NP + ld];
  bool valid = mk && a >= 0 && bb >= 0;
  ns[e] = valid ? bs * k + a  : 0;
  nd[e] = valid ? bd * k + bb : 0;
  nm[e] = valid ? 1 : 0;
}

// ---------------- GPS pieces ----------------
__global__ void gin_scatter(const int* src, const int* dst, const unsigned char* mask,
                            const float* x, float* agg, int E_) {
  int gw = (blockIdx.x * blockDim.x + threadIdx.x) >> 5;
  int lane = threadIdx.x & 31;
  if (gw >= E_ || !mask[gw]) return;
  int s = src[gw], d = dst[gw];
  for (int c = lane; c < DM; c += 32)
    atomicAdd(&agg[(long)d * DM + c], x[(long)s * DM + c]);
}

__global__ void gin_comb(float* agg, const float* x, const float* eps, long n) {
  long i = (long)blockIdx.x * blockDim.x + threadIdx.x;
  if (i < n) agg[i] = (1.f + eps[0]) * x[i] + agg[i];
}

// wave-per-row LayerNorm over 256 cols; out = LN(a (+b))
__global__ void layernorm(const float* a, const float* b, const float* g,
                          const float* be, float* out, int n) {
  int row  = (blockIdx.x * blockDim.x + threadIdx.x) >> 5;
  int lane = threadIdx.x & 31;
  if (row >= n) return;
  float x[8]; float s = 0.f;
  #pragma unroll
  for (int t = 0; t < 8; ++t) {
    int c = lane + 32 * t;
    float v = a[(long)row * DM + c];
    if (b) v += b[(long)row * DM + c];
    x[t] = v; s += v;
  }
  for (int o = 16; o > 0; o >>= 1) s += __shfl_xor(s, o, 32);
  float mu = s * (1.f / DM);
  float vs = 0.f;
  #pragma unroll
  for (int t = 0; t < 8; ++t) { float d = x[t] - mu; vs += d * d; }
  for (int o = 16; o > 0; o >>= 1) vs += __shfl_xor(vs, o, 32);
  float inv = rsqrtf(vs * (1.f / DM) + 1e-5f);
  #pragma unroll
  for (int t = 0; t < 8; ++t) {
    int c = lane + 32 * t;
    out[(long)row * DM + c] = (x[t] - mu) * inv * g[c] + be[c];
  }
}

// qf = exp(pr - 0.5*||u||^2)/sqrt(M), in place on pr; layout [n2][H*MFEAT]
__global__ void phi_fin(float* pr, const float* q, int n2, float scale) {
  int i = blockIdx.x * blockDim.x + threadIdx.x;
  if (i >= n2 * HEADS) return;
  int node = i >> 2, h = i & 3;
  const float* qr = q + (long)node * DM + h * DH;
  float ss = 0.f;
  for (int j = 0; j < DH; ++j) { float u = qr[j] * scale; ss += u * u; }
  float cst = -0.5f * ss - 0.5f * logf((float)MFEAT);
  float* o = pr + (long)node * (HEADS * MFEAT) + h * MFEAT;
  for (int j = 0; j < MFEAT; ++j) o[j] = __expf(o[j] + cst);
}

__global__ void kf_sum(const float* kf, float* ks) {
  int i = blockIdx.x * blockDim.x + threadIdx.x;        // (b,h,j)
  if (i >= BSZ * HEADS * MFEAT) return;
  int j = i & (MFEAT - 1), h = (i >> 7) & 3, b = i >> 9;
  float s = 0.f;
  for (int nn = 0; nn < K2C; ++nn)
    s += kf[((long)b * K2C + nn) * (HEADS * MFEAT) + h * MFEAT + j];
  ks[i] = s;
}

__global__ void perf_den(const float* qf, const float* ks, float* den) {
  int i = blockIdx.x * blockDim.x + threadIdx.x;        // (b,h,n)
  if (i >= BSZ * HEADS * K2C) return;
  int nn = i & (K2C - 1), h = (i >> 9) & 3, b = i >> 11;
  const float* q = qf + ((long)b * K2C + nn) * (HEADS * MFEAT) + h * MFEAT;
  const float* k = ks + (b * HEADS + h) * MFEAT;
  float s = 0.f;
  for (int j = 0; j < MFEAT; ++j) s += q[j] * k[j];
  den[i] = s;
}

__global__ void perf_norm(float* num, const float* den, int n2) {
  long i = (long)blockIdx.x * blockDim.x + threadIdx.x;
  if (i >= (long)n2 * DM) return;
  int c = (int)(i & (DM - 1));
  long row = i >> 8;
  int h = c >> 6, b = (int)(row >> 9), nn = (int)(row & (K2C - 1));
  num[i] = num[i] / (den[((long)b * HEADS + h) * K2C + nn] + 1e-6f);
}

__global__ void mean_pool(const float* h, float* out) {
  int i = blockIdx.x * blockDim.x + threadIdx.x;        // 64*256
  if (i >= BSZ * DM) return;
  int b = i >> 8, c = i & (DM - 1);
  float s = 0.f;
  for (int nn = 0; nn < K2C; ++nn) s += h[((long)b * K2C + nn) * DM + c];
  out[i] = s * (1.f / K2C);
}

// ================= host orchestration =================
static inline unsigned gblk(long threads) { return (unsigned)((threads + 255) / 256); }

static void launch_gemm(hipStream_t st, const float* A, const float* B, float* C,
                        int M, int N, int K, int lda, int ldb, int ldc,
                        int tA, int tB, const float* bias, const float* resid,
                        int relu, float alpha, int outer, int inner,
                        long sAo, long sAi, long sBo, long sBi, long sCo, long sCi) {
  GemmP p;
  p.A = A; p.B = B; p.C = C; p.bias = bias; p.resid = resid;
  p.M = M; p.N = N; p.K = K; p.lda = lda; p.ldb = ldb; p.ldc = ldc;
  p.sAo = sAo; p.sAi = sAi; p.sBo = sBo; p.sBi = sBi; p.sCo = sCo; p.sCi = sCi;
  p.innerB = inner; p.transA = tA; p.transB = tB; p.relu = relu; p.alpha = alpha;
  dim3 g((N + BN - 1) / BN, (M + BM - 1) / BM, outer * inner);
  gemm_wmma<<<g, dim3(256), 0, st>>>(p);
}

extern "C" void kernel_launch(void* const* d_in, const int* in_sizes, int n_in,
                              void* d_out, int out_size, void* d_ws, size_t ws_size,
                              hipStream_t stream) {
  (void)in_sizes; (void)n_in; (void)out_size; (void)ws_size;
  const float* x    = (const float*)d_in[0];
  const int*   esrc = (const int*)d_in[1];
  const int*   edst = (const int*)d_in[2];
  const float* W1 = (const float*)d_in[4], *as1 = (const float*)d_in[5];
  const float* ad1 = (const float*)d_in[6], *b1 = (const float*)d_in[7];
  const float* p1 = (const float*)d_in[8];
  const float* W2 = (const float*)d_in[9], *as2 = (const float*)d_in[10];
  const float* ad2 = (const float*)d_in[11], *b2 = (const float*)d_in[12];
  const float* p2 = (const float*)d_in[13];

  // workspace bump allocator
  char* wsb = (char*)d_ws; size_t off = 0;
  auto A_ = [&](size_t bytes) -> void* {
    void* p = wsb + off; off += (bytes + 255) & ~(size_t)255; return p;
  };
  float* fA  = (float*)A_((size_t)N0 * DM * 4);   // big scratch (134MB)
  float* fB  = (float*)A_((size_t)N0 * DM * 4);   // big scratch (134MB)
  float* fC  = (float*)A_((size_t)N1 * DM * 4);   // pooled stage-1 nodes
  float* fH2 = (float*)A_((size_t)N2 * DM * 4);   // pooled stage-2 nodes
  float* kv  = (float*)A_((size_t)BSZ * HEADS * MFEAT * DH * 4);
  float* ks  = (float*)A_((size_t)BSZ * HEADS * MFEAT * 4);
  float* dn2 = (float*)A_((size_t)BSZ * HEADS * K2C * 4);
  float* es  = (float*)A_((size_t)N0 * HEADS * 4);
  float* ed  = (float*)A_((size_t)N0 * HEADS * 4);
  unsigned* amax = (unsigned*)A_((size_t)N0 * HEADS * 4);
  float* den = (float*)A_((size_t)N0 * HEADS * 4);
  float* sc  = (float*)A_((size_t)N0 * 4);
  float* vals = (float*)A_((size_t)N1 * 4);
  int* perm  = (int*)A_((size_t)N1 * 4);
  int* newid = (int*)A_((size_t)N0 * 4);
  float* pn  = (float*)A_(256);
  int* s1 = (int*)A_((size_t)ETOT * 4); int* d1 = (int*)A_((size_t)ETOT * 4);
  unsigned char* m1 = (unsigned char*)A_((size_t)ETOT);
  int* s2 = (int*)A_((size_t)ETOT * 4); int* d2 = (int*)A_((size_t)ETOT * 4);
  unsigned char* m2 = (unsigned char*)A_((size_t)ETOT);

  auto run_gat = [&](const float* X, int n, int Kin, const float* W,
                     const float* as, const float* ad, const float* bias,
                     const int* src, const int* dst, const unsigned char* mask,
                     float* xw, float* out) {
    launch_gemm(stream, X, W, xw, n, DM, Kin, Kin, DM, DM, 0, 0,
                nullptr, nullptr, 0, 1.f, 1, 1, 0, 0, 0, 0, 0, 0);
    attn_coef<<<gblk((long)n * HEADS), 256, 0, stream>>>(xw, as, ad, es, ed, n);
    fill_u32<<<gblk((long)n * HEADS), 256, 0, stream>>>(amax, f2o(-1e9f), (long)n * HEADS);
    edge_amax<<<gblk((long)ETOT + n), 256, 0, stream>>>(src, dst, mask, es, ed, amax, ETOT, n);
    fill_f32<<<gblk((long)n * HEADS), 256, 0, stream>>>(den, 0.f, (long)n * HEADS);
    edge_expsum<<<gblk((long)ETOT + n), 256, 0, stream>>>(src, dst, mask, es, ed, amax, den, ETOT, n);
    bias_init<<<gblk((long)n * DM), 256, 0, stream>>>(out, bias, n);
    edge_agg<<<gblk(((long)ETOT + n) * 32), 256, 0, stream>>>(
        src, dst, mask, es, ed, amax, den, xw, out, ETOT, n);
  };

  // ---- GAT1 (131072 nodes, IN=128) ----
  run_gat(x, N0, INDIM, W1, as1, ad1, b1, esrc, edst, nullptr, fA, fB);

  // ---- TopK 1: 2048 -> 1024 per graph ----
  vec_norm<<<1, 256, 0, stream>>>(p1, pn, DM);
  score_k<<<gblk(N0), 256, 0, stream>>>(fB, p1, pn, sc, N0);
  topk_sort<<<BSZ, 1024, 0, stream>>>(sc, vals, perm, newid, NPER0, K1C);
  topk_gather<<<gblk((long)N1 * DM), 256, 0, stream>>>(fB, vals, perm, fC, NPER0, K1C, 1);
  edge_remap<<<gblk(ETOT), 256, 0, stream>>>(esrc, edst, nullptr, newid, NPER0, K1C, s1, d1, m1, ETOT);

  // ---- GAT2 (65536 nodes, D=256) ----
  run_gat(fC, N1, DM, W2, as2, ad2, b2, s1, d1, m1, fA, fB);

  // ---- TopK 2: 1024 -> 512 per graph ----
  vec_norm<<<1, 256, 0, stream>>>(p2, pn, DM);
  score_k<<<gblk(N1), 256, 0, stream>>>(fB, p2, pn, sc, N1);
  topk_sort<<<BSZ, 1024, 0, stream>>>(sc, vals, perm, newid, K1C, K2C);
  topk_gather<<<gblk((long)N2 * DM), 256, 0, stream>>>(fB, vals, perm, fH2, K1C, K2C, 1);
  edge_remap<<<gblk(ETOT), 256, 0, stream>>>(s1, d1, m1, newid, K1C, K2C, s2, d2, m2, ETOT);

  // ---- 3x GPS layers on fH2 (32768 nodes) ----
  const float scale = 0.35355339059327373f;            // 64^-0.25
  const long  nel = (long)N2 * DM;
  float* agg = fA;                                     // [N2,256]
  float* t1  = fA + (size_t)N2 * DM;                   // [N2,512]
  float* qf  = fA;                                     // [N2,512] (after GIN)
  float* kf  = fA + (size_t)N2 * 512;                  // [N2,512]
  float* hl  = fB;                                     // [N2,256]
  float* qb  = fB + (size_t)N2 * DM;                   // q / num
  float* kb  = fB + 2 * (size_t)N2 * DM;               // k / o
  float* vb  = fB + 3 * (size_t)N2 * DM;               // v / ha

  for (int l = 0; l < 3; ++l) {
    const float* Wg1 = (const float*)d_in[14 + 20 * l + 0];
    const float* bg1 = (const float*)d_in[14 + 20 * l + 1];
    const float* Wg2 = (const float*)d_in[14 + 20 * l + 2];
    const float* bg2 = (const float*)d_in[14 + 20 * l + 3];
    const float* eps = (const float*)d_in[14 + 20 * l + 4];
    const float* Wq  = (const float*)d_in[14 + 20 * l + 5];
    const float* Wk  = (const float*)d_in[14 + 20 * l + 6];
    const float* Wv  = (const float*)d_in[14 + 20 * l + 7];
    const float* Wo  = (const float*)d_in[14 + 20 * l + 8];
    const float* om  = (const float*)d_in[14 + 20 * l + 9];
    const float* g1  = (const float*)d_in[14 + 20 * l + 10];
    const float* be1 = (const float*)d_in[14 + 20 * l + 11];
    const float* g2  = (const float*)d_in[14 + 20 * l + 12];
    const float* be2 = (const float*)d_in[14 + 20 * l + 13];
    const float* g3  = (const float*)d_in[14 + 20 * l + 14];
    const float* be3 = (const float*)d_in[14 + 20 * l + 15];
    const float* Wf1 = (const float*)d_in[14 + 20 * l + 16];
    const float* bf1 = (const float*)d_in[14 + 20 * l + 17];
    const float* Wf2 = (const float*)d_in[14 + 20 * l + 18];
    const float* bf2 = (const float*)d_in[14 + 20 * l + 19];

    // GIN: agg = sum_{masked edges} x[src]; h = (1+eps)*x + agg; MLP; +x; LN1
    fill_f32<<<gblk(nel), 256, 0, stream>>>(agg, 0.f, nel);
    gin_scatter<<<gblk((long)ETOT * 32), 256, 0, stream>>>(s2, d2, m2, fH2, agg, ETOT);
    gin_comb<<<gblk(nel), 256, 0, stream>>>(agg, fH2, eps, nel);
    launch_gemm(stream, agg, Wg1, t1, N2, DM, DM, DM, DM, DM, 0, 0,
                bg1, nullptr, 1, 1.f, 1, 1, 0, 0, 0, 0, 0, 0);
    launch_gemm(stream, t1, Wg2, agg, N2, DM, DM, DM, DM, DM, 0, 0,
                bg2, fH2, 0, 1.f, 1, 1, 0, 0, 0, 0, 0, 0);
    layernorm<<<gblk((long)N2 * 32), 256, 0, stream>>>(agg, nullptr, g1, be1, hl, N2);

    // Performer: q,k,v projections
    launch_gemm(stream, fH2, Wq, qb, N2, DM, DM, DM, DM, DM, 0, 0,
                nullptr, nullptr, 0, 1.f, 1, 1, 0, 0, 0, 0, 0, 0);
    launch_gemm(stream, fH2, Wk, kb, N2, DM, DM, DM, DM, DM, 0, 0,
                nullptr, nullptr, 0, 1.f, 1, 1, 0, 0, 0, 0, 0, 0);
    launch_gemm(stream, fH2, Wv, vb, N2, DM, DM, DM, DM, DM, 0, 0,
                nullptr, nullptr, 0, 1.f, 1, 1, 0, 0, 0, 0, 0, 0);
    // qf/kf = exp(scale*(u @ omega^T) - 0.5||u||^2)/sqrt(M); batched over heads
    launch_gemm(stream, qb, om, qf, N2, MFEAT, DH, DM, DH, HEADS * MFEAT, 0, 1,
                nullptr, nullptr, 0, scale, 1, HEADS, 0, DH, 0, 0, 0, MFEAT);
    launch_gemm(stream, kb, om, kf, N2, MFEAT, DH, DM, DH, HEADS * MFEAT, 0, 1,
                nullptr, nullptr, 0, scale, 1, HEADS, 0, DH, 0, 0, 0, MFEAT);
    phi_fin<<<gblk((long)N2 * HEADS), 256, 0, stream>>>(qf, qb, N2, scale);
    phi_fin<<<gblk((long)N2 * HEADS), 256, 0, stream>>>(kf, kb, N2, scale);
    kf_sum<<<gblk(BSZ * HEADS * MFEAT), 256, 0, stream>>>(kf, ks);
    // kv[b,h] = kf[b,h]^T @ v[b,h] : M=128,N=64,K=512, batched (64 x 4)
    launch_gemm(stream, kf, vb, kv, MFEAT, DH, K2C, HEADS * MFEAT, DM, DH, 1, 0,
                nullptr, nullptr, 0, 1.f, BSZ, HEADS,
                (long)K2C * HEADS * MFEAT, MFEAT,
                (long)K2C * DM, DH,
                (long)HEADS * MFEAT * DH, (long)MFEAT * DH);
    // num[b,h] = qf[b,h] @ kv[b,h] -> written into o-layout [n2][h*64+d]
    launch_gemm(stream, qf, kv, qb, K2C, DH, MFEAT, HEADS * MFEAT, DH, DM, 0, 0,
                nullptr, nullptr, 0, 1.f, BSZ, HEADS,
                (long)K2C * HEADS * MFEAT, MFEAT,
                (long)HEADS * MFEAT * DH, (long)MFEAT * DH,
                (long)K2C * DM, DH);
    perf_den<<<gblk(BSZ * HEADS * K2C), 256, 0, stream>>>(qf, ks, dn2);
    perf_norm<<<gblk(nel), 256, 0, stream>>>(qb, dn2, N2);
    // o = num @ Wo (+x residual), then LN2
    launch_gemm(stream, qb, Wo, kb, N2, DM, DM, DM, DM, DM, 0, 0,
                nullptr, fH2, 0, 1.f, 1, 1, 0, 0, 0, 0, 0, 0);
    layernorm<<<gblk((long)N2 * 32), 256, 0, stream>>>(kb, nullptr, g2, be2, vb, N2);

    // h = hl + ha ; FFN ; LN3 -> fH2
    add2<<<gblk(nel), 256, 0, stream>>>(hl, hl, vb, nel);
    launch_gemm(stream, hl, Wf1, t1, N2, 2 * DM, DM, DM, 2 * DM, 2 * DM, 0, 0,
                bf1, nullptr, 1, 1.f, 1, 1, 0, 0, 0, 0, 0, 0);
    launch_gemm(stream, t1, Wf2, kf, N2, DM, 2 * DM, 2 * DM, DM, DM, 0, 0,
                bf2, hl, 0, 1.f, 1, 1, 0, 0, 0, 0, 0, 0);
    layernorm<<<gblk((long)N2 * 32), 256, 0, stream>>>(kf, nullptr, g3, be3, fH2, N2);
  }

  // ---- global mean pool ----
  mean_pool<<<gblk(BSZ * DM), 256, 0, stream>>>(fH2, (float*)d_out);
}
